// LSTM_CRF_80169859547343
// MI455X (gfx1250) — compile-verified, matrix operand order checked by reading, not run on hardware
//
#include <hip/hip_runtime.h>

// Problem constants
constexpr int BATCH = 128;   // B
constexpr int TT    = 256;   // T
constexpr int HH    = 256;   // H
constexpr int EE    = 300;   // E
constexpr int EPAD  = 320;   // E padded to multiple of 32 (K of x@W)
constexpr int NTAG  = 9;     // CRF tags
constexpr int FOURH = 1024;  // 4H
constexpr int CATD  = 512;   // 2H (concat width)

typedef __attribute__((ext_vector_type(16))) __bf16       v16bf;
typedef __attribute__((ext_vector_type(8)))  float        v8f;
typedef __attribute__((ext_vector_type(4)))  unsigned int v4u;
typedef __attribute__((ext_vector_type(8)))  unsigned int v8u;

// ---- helpers -------------------------------------------------------------

__device__ __forceinline__ unsigned short f2bf(float f) {
    union { float f; unsigned u; } x; x.f = f;
    unsigned r = x.u + 0x7FFFu + ((x.u >> 16) & 1u);   // round-to-nearest-even
    return (unsigned short)(r >> 16);
}

// Load one lane's 16-element bf16 fragment: K chunks [k0..k0+7] and [k0+16..k0+23]
// (matches CDNA5 16-bit A/B matrix VGPR layout) as two global_load_b128.
__device__ __forceinline__ v16bf load_frag(const unsigned short* p) {
    v4u lo = *(const v4u*)(p);
    v4u hi = *(const v4u*)(p + 16);
    v8u c;
    c[0] = lo[0]; c[1] = lo[1]; c[2] = lo[2]; c[3] = lo[3];
    c[4] = hi[0]; c[5] = hi[1]; c[6] = hi[2]; c[7] = hi[3];
    return __builtin_bit_cast(v16bf, c);
}

__device__ __forceinline__ float sigmoidf_(float x) { return 1.0f / (1.0f + __expf(-x)); }

// ---- setup kernels -------------------------------------------------------

// zero h ping-pong buffers (bf16) and cell state (f32)
__global__ void k_init(unsigned short* hprev, float* cst) {
    int i = blockIdx.x * blockDim.x + threadIdx.x;
    if (i < 4 * BATCH * HH) hprev[i] = 0;
    if (i < 2 * BATCH * HH) cst[i] = 0.0f;
}

// W [E,4H] f32 -> Wt [dir][4H][EPAD] bf16 (transposed + K-padded)
__global__ void k_packW(const float* __restrict__ Wf, const float* __restrict__ Wb,
                        unsigned short* __restrict__ Wt) {
    int i = blockIdx.x * blockDim.x + threadIdx.x;
    if (i >= 2 * FOURH * EPAD) return;
    int k = i % EPAD; int n = (i / EPAD) % FOURH; int dir = i / (EPAD * FOURH);
    const float* W = dir ? Wb : Wf;
    float v = (k < EE) ? W[(size_t)k * FOURH + n] : 0.0f;
    Wt[i] = f2bf(v);
}

// U [H,4H] f32 -> Ut [dir][4H][H] bf16 (transposed)
__global__ void k_packU(const float* __restrict__ Uf, const float* __restrict__ Ub,
                        unsigned short* __restrict__ Ut) {
    int i = blockIdx.x * blockDim.x + threadIdx.x;
    if (i >= 2 * FOURH * HH) return;
    int k = i % HH; int n = (i / HH) % FOURH; int dir = i / (HH * FOURH);
    const float* U = dir ? Ub : Uf;
    Ut[i] = f2bf(U[(size_t)k * FOURH + n]);
}

// fcW [2H,9] f32 -> fcWt [16][2H] bf16 (transposed, N padded to 16)
__global__ void k_packFC(const float* __restrict__ fcW, unsigned short* __restrict__ fcWt) {
    int i = blockIdx.x * blockDim.x + threadIdx.x;
    if (i >= 16 * CATD) return;
    int k = i % CATD; int n = i / CATD;
    float v = (n < NTAG) ? fcW[(size_t)k * NTAG + n] : 0.0f;
    fcWt[i] = f2bf(v);
}

// text_lens = count of nonzero tokens per row
__global__ void k_lens(const int* __restrict__ inputs, int* __restrict__ lens_i,
                       float* __restrict__ out_lens) {
    __shared__ int sm[TT];
    int b = blockIdx.x, t = threadIdx.x;
    sm[t] = (inputs[b * TT + t] != 0) ? 1 : 0;
    __syncthreads();
    for (int s = TT / 2; s > 0; s >>= 1) {
        if (t < s) sm[t] += sm[t + s];
        __syncthreads();
    }
    if (t == 0) { lens_i[b] = sm[0]; out_lens[b] = (float)sm[0]; }
}

// x[bt][e] = bf16(emb[tok][e]) with zero padding e>=E
__global__ void k_embed(const int* __restrict__ inputs, const float* __restrict__ emb,
                        unsigned short* __restrict__ x) {
    int bt = blockIdx.x, e = threadIdx.x;
    int tok = inputs[bt];
    float v = (e < EE) ? emb[(size_t)tok * EE + e] : 0.0f;
    x[(size_t)bt * EPAD + e] = f2bf(v);
}

// ---- LSTM step: z = x_t@W + h_{t-1}@U + b, gate update, emit h ----------
// grid (128, 2): blockIdx.x = btile*16 + htile, blockIdx.y = direction
// block 128 = 4 waves, wave g computes the 16x16 z-tile of gate g.
__global__ void __launch_bounds__(128) k_step(
    int t,
    const unsigned short* __restrict__ x,     // [B*T][EPAD] bf16
    const unsigned short* __restrict__ Wt,    // [dir][4H][EPAD] bf16
    const unsigned short* __restrict__ Ut,    // [dir][4H][H] bf16
    const float* __restrict__ bfv, const float* __restrict__ bbv,
    unsigned short* __restrict__ hprev,       // [2 buf][dir][B][H] bf16 ping-pong
    float* __restrict__ cst,                  // [dir][B][H] f32
    unsigned short* __restrict__ hcat)        // [B][T][2H] bf16
{
    const int dir   = blockIdx.y;
    const int btile = blockIdx.x >> 4;
    const int htile = blockIdx.x & 15;
    const int lane  = threadIdx.x & 31;
    const int wave  = threadIdx.x >> 5;
    const int lm    = lane & 15;
    const int kh    = (lane >> 4) << 3;          // 0 or 8
    const int t_mem = dir ? (TT - 1 - t) : t;
    const int M0    = btile * 16;
    const int N0    = wave * HH + htile * 16;    // column in 4H (gate = wave)

    v8f acc = {0.f, 0.f, 0.f, 0.f, 0.f, 0.f, 0.f, 0.f};

    // x_t @ W : K = EPAD = 320 -> 10 WMMAs
    const unsigned short* xrow = x  + ((size_t)(M0 + lm) * TT + t_mem) * EPAD;
    const unsigned short* wrow = Wt + (size_t)(dir * FOURH + N0 + lm) * EPAD;
#pragma unroll
    for (int ks = 0; ks < EPAD / 32; ++ks) {
        int k0 = ks * 32 + kh;
        v16bf a = load_frag(xrow + k0);
        v16bf b = load_frag(wrow + k0);
        acc = __builtin_amdgcn_wmma_f32_16x16x32_bf16(false, a, false, b,
                                                      (short)0, acc, false, false);
    }
    // h_{t-1} @ U : K = H = 256 -> 8 WMMAs (read ping buffer t&1)
    const unsigned short* hrow = hprev + (size_t)(((t & 1) * 2 + dir) * BATCH + M0 + lm) * HH;
    const unsigned short* urow = Ut + (size_t)(dir * FOURH + N0 + lm) * HH;
#pragma unroll
    for (int ks = 0; ks < HH / 32; ++ks) {
        int k0 = ks * 32 + kh;
        v16bf a = load_frag(hrow + k0);
        v16bf b = load_frag(urow + k0);
        acc = __builtin_amdgcn_wmma_f32_16x16x32_bf16(false, a, false, b,
                                                      (short)0, acc, false, false);
    }
    const float bias = (dir ? bbv : bfv)[N0 + lm];

    // Stage the 4 gate tiles in LDS: zs[gate][m][n]
    __shared__ float zs[4 * 256];
#pragma unroll
    for (int r = 0; r < 8; ++r)
        zs[wave * 256 + (r + kh) * 16 + lm] = acc[r] + bias;   // C/D layout: m = r + 8*(lane>=16), n = lm
    __syncthreads();

    // Elementwise gate update: 256 elements, 2 per thread
    for (int e = threadIdx.x; e < 256; e += 128) {
        int m = e >> 4, n = e & 15;
        float ig = sigmoidf_(zs[e]);
        float fg = sigmoidf_(zs[256 + e]);
        float gg = tanhf(zs[512 + e]);
        float og = sigmoidf_(zs[768 + e]);
        int b  = M0 + m;
        int hc = htile * 16 + n;
        size_t ci = (size_t)(dir * BATCH + b) * HH + hc;
        float cc = fg * cst[ci] + ig * gg;
        cst[ci] = cc;
        float hh = og * tanhf(cc);
        unsigned short hb = f2bf(hh);
        hprev[(size_t)((((t + 1) & 1) * 2 + dir) * BATCH + b) * HH + hc] = hb;  // pong
        hcat[((size_t)(b * TT + t_mem)) * CATD + dir * HH + hc] = hb;
    }
}

// ---- logits = hcat @ fcW + fcb via WMMA (N padded 9->16, K=512) ---------
__global__ void __launch_bounds__(256) k_logits(
    const unsigned short* __restrict__ hcat,   // [B*T][512] bf16
    const unsigned short* __restrict__ fcWt,   // [16][512] bf16
    const float* __restrict__ fcb,
    float* __restrict__ out)                   // [B*T][9] f32
{
    const int wid  = blockIdx.x * 8 + (threadIdx.x >> 5);   // 2048 M-tiles
    const int lane = threadIdx.x & 31;
    const int lm   = lane & 15;
    const int kh   = (lane >> 4) << 3;
    const int M0   = wid * 16;

    v8f acc = {0.f, 0.f, 0.f, 0.f, 0.f, 0.f, 0.f, 0.f};
    const unsigned short* arow = hcat + (size_t)(M0 + lm) * CATD;
    const unsigned short* brow = fcWt + (size_t)lm * CATD;
#pragma unroll
    for (int ks = 0; ks < CATD / 32; ++ks) {
        int k0 = ks * 32 + kh;
        v16bf a = load_frag(arow + k0);
        v16bf b = load_frag(brow + k0);
        acc = __builtin_amdgcn_wmma_f32_16x16x32_bf16(false, a, false, b,
                                                      (short)0, acc, false, false);
    }
#pragma unroll
    for (int r = 0; r < 8; ++r) {
        int m = r + kh, n = lm;
        if (n < NTAG) out[(size_t)(M0 + m) * NTAG + n] = acc[r] + fcb[n];
    }
}

// ---- CRF log-likelihood: one wave per batch row, lane j = tag j ---------
__global__ void k_crf(const float* __restrict__ logits, const int* __restrict__ labels,
                      const int* __restrict__ lens_i, const float* __restrict__ trans,
                      float* __restrict__ ll)
{
    const int b    = blockIdx.x;
    const int lane = threadIdx.x;
    const int jj   = lane < NTAG ? lane : NTAG - 1;
    const int len  = lens_i[b];
    const float* lg = logits + (size_t)b * TT * NTAG;
    const int*   lab = labels + b * TT;

    float tr[NTAG];
#pragma unroll
    for (int i = 0; i < NTAG; ++i) tr[i] = trans[i * NTAG + jj];

    // gold path score (uniform across lanes)
    float score = lg[lab[0]];
    int prev = lab[0];
    for (int t = 1; t < TT; ++t) {
        int l = lab[t];
        if (t < len) score += lg[t * NTAG + l] + trans[prev * NTAG + l];
        prev = l;
    }
    // forward algorithm: lane j holds alpha[j]
    float alpha = lg[jj];
    for (int t = 1; t < TT; ++t) {
        float lt = lg[t * NTAG + jj];
        float av[NTAG]; float m = -3.0e38f;
#pragma unroll
        for (int i = 0; i < NTAG; ++i) {
            av[i] = __shfl(alpha, i, 32) + tr[i];
            m = fmaxf(m, av[i]);
        }
        float s = 0.f;
#pragma unroll
        for (int i = 0; i < NTAG; ++i) s += __expf(av[i] - m);
        float nw = m + __logf(s) + lt;
        alpha = (t < len) ? nw : alpha;
    }
    // log-partition = LSE over tags
    float am[NTAG]; float m2 = -3.0e38f;
#pragma unroll
    for (int i = 0; i < NTAG; ++i) {
        am[i] = __shfl(alpha, i, 32);
        m2 = fmaxf(m2, am[i]);
    }
    float s2 = 0.f;
#pragma unroll
    for (int i = 0; i < NTAG; ++i) s2 += __expf(am[i] - m2);
    if (lane == 0) ll[b] = score - (m2 + __logf(s2));
}

// ---- launch --------------------------------------------------------------

extern "C" void kernel_launch(void* const* d_in, const int* in_sizes, int n_in,
                              void* d_out, int out_size, void* d_ws, size_t ws_size,
                              hipStream_t stream)
{
    const int*   inputs = (const int*)d_in[0];
    const int*   labels = (const int*)d_in[1];
    const float* emb    = (const float*)d_in[2];
    const float* Wf     = (const float*)d_in[3];
    const float* Uf     = (const float*)d_in[4];
    const float* bfv    = (const float*)d_in[5];
    const float* Wb     = (const float*)d_in[6];
    const float* Ub     = (const float*)d_in[7];
    const float* bbv    = (const float*)d_in[8];
    const float* fcW    = (const float*)d_in[9];
    const float* fcb    = (const float*)d_in[10];
    const float* trans  = (const float*)d_in[11];

    // workspace layout (bytes)
    char* ws = (char*)d_ws;
    unsigned short* x     = (unsigned short*)(ws + 0);          // 32768*320*2  = 20,971,520
    unsigned short* Wt    = (unsigned short*)(ws + 20971520);   // 2*1024*320*2 =  1,310,720
    unsigned short* Ut    = (unsigned short*)(ws + 22282240);   // 2*1024*256*2 =  1,048,576
    unsigned short* fct   = (unsigned short*)(ws + 23330816);   // 16*512*2     =     16,384
    unsigned short* hcat  = (unsigned short*)(ws + 23347200);   // 32768*512*2  = 33,554,432
    unsigned short* hprev = (unsigned short*)(ws + 56901632);   // 4*128*256*2  =    262,144
    float*          cst   = (float*)         (ws + 57163776);   // 2*128*256*4  =    262,144
    int*            lens_i= (int*)           (ws + 57425920);   // 128*4

    float* out_logits = (float*)d_out;                  // B*T*NTAG = 294912
    float* out_ll     = out_logits + BATCH * TT * NTAG; // 128
    float* out_lens   = out_ll + BATCH;                 // 128

    k_init  <<<dim3((4 * BATCH * HH + 255) / 256), 256, 0, stream>>>(hprev, cst);
    k_packW <<<dim3((2 * FOURH * EPAD + 255) / 256), 256, 0, stream>>>(Wf, Wb, Wt);
    k_packU <<<dim3((2 * FOURH * HH + 255) / 256), 256, 0, stream>>>(Uf, Ub, Ut);
    k_packFC<<<dim3((16 * CATD + 255) / 256), 256, 0, stream>>>(fcW, fct);
    k_lens  <<<dim3(BATCH), TT, 0, stream>>>(inputs, lens_i, out_lens);
    k_embed <<<dim3(BATCH * TT), EPAD, 0, stream>>>(inputs, emb, x);

    for (int t = 0; t < TT; ++t)
        k_step<<<dim3(128, 2), 128, 0, stream>>>(t, x, Wt, Ut, bfv, bbv, hprev, cst, hcat);

    k_logits<<<dim3(256), 256, 0, stream>>>(hcat, fct, fcb, out_logits);
    k_crf   <<<dim3(BATCH), 32, 0, stream>>>(out_logits, labels, lens_i, trans, out_ll);
}
